// Criterion_28003186770325
// MI455X (gfx1250) — matrix operand order repaired and checked by reading, not compiled, not run on hardware
//
#include <hip/hip_runtime.h>
#include <hip/hip_bf16.h>

// ---------------------------------------------------------------------------
// Problem constants (from the reference)
// ---------------------------------------------------------------------------
#define BB   8
#define VV   1024
#define DD   128
#define CC   12
#define HH   224
#define WW   224
#define MROWS   (BB * VV)        // 8192
#define NCOLS   (CC * VV)        // 12288
#define MT      (MROWS / 16)     // 512 row tiles
#define NT      (NCOLS / 16)     // 768 col tiles
#define KB      (DD / 32)        // 4 k-blocks of 32
#define CHUNKS  4
#define TPC     (NT / CHUNKS)    // 192 col tiles per chunk
#define KAPPA   14.2857142857f   // 1 / 0.07
#define NEG_INF_F (-1.0e30f)
#define PAD_BIAS  (-1.4285714e31f)   // NEG_INF_F * KAPPA
#define BG_N    (BB * HH * WW)   // 401408

typedef __attribute__((ext_vector_type(16))) __bf16 v16bf;
typedef __attribute__((ext_vector_type(8)))  __bf16 v8bf;
typedef __attribute__((ext_vector_type(8)))  float  v8f;
typedef unsigned int u32x4 __attribute__((ext_vector_type(4)));
typedef int          i32x4 __attribute__((ext_vector_type(4)));
typedef int          i32x8 __attribute__((ext_vector_type(8)));

#define SHUF16(lo, hh) \
    __builtin_shufflevector(lo, hh, 0,1,2,3,4,5,6,7,8,9,10,11,12,13,14,15)

// ---------------------------------------------------------------------------
// Workspace layout (bytes):
//   [0,64)            : float accum[16] (0:ce_m 1:vis_m 2:ce_i 3:vis_i 4:bg_m 5:bg_i)
//   [64, 49216)       : rnorm[12288]
//   [64K, +256K)      : Pm partial max   [set][chunk][row]
//   [64K+256K, +256K) : Ps partial sumexp
//   [64K+512K, +256K) : Pt partial target logit
//   [1M, +2M)         : A_m packed bf16 fragments
//   [3M, +2M)         : A_i packed bf16 fragments
//   [5M, +3M)         : B packed bf16 fragments (tile = 4KB contiguous)
// ---------------------------------------------------------------------------

__global__ void k_init_accum(float* accum) {
    if (threadIdx.x < 16) accum[threadIdx.x] = 0.0f;
}

// One wave per (c,j) row: rnorm = 1/||nmm[c,j,:]||
__global__ __launch_bounds__(256) void k_norm(const float* __restrict__ nmm,
                                              float* __restrict__ rnorm) {
    int gw   = (blockIdx.x * blockDim.x + threadIdx.x) >> 5;
    int lane = threadIdx.x & 31;
    if (gw >= NCOLS) return;
    const float4* p = (const float4*)(nmm + (size_t)gw * DD);
    float4 v = p[lane];
    float ss = v.x * v.x + v.y * v.y + v.z * v.z + v.w * v.w;
    #pragma unroll
    for (int mask = 1; mask < 32; mask <<= 1) ss += __shfl_xor(ss, mask, 32);
    if (lane == 0) rnorm[gw] = rsqrtf(ss);
}

// Pack A into WMMA A-fragments (16-bit A 16x32, contiguous k: lanes0-15 K=0..15, +16 hi; same order as B).
__global__ __launch_bounds__(256) void k_packA(const float* __restrict__ km,
                                               const float* __restrict__ ki,
                                               v16bf* __restrict__ Am,
                                               v16bf* __restrict__ Ai) {
    int tid = blockIdx.x * blockDim.x + threadIdx.x;
    int set     = tid >> 16;
    int x       = tid & 65535;
    int fragidx = x >> 5;                                  // mt*4 + kb
    int lane    = x & 31;
    int mt = fragidx >> 2, kb = fragidx & 3;
    int hi = lane >> 4;
    int row = mt * 16 + (lane & 15);
    const float* src = (set ? ki : km) + (size_t)row * DD + kb * 32 + hi * 16;
    v16bf frag;
    #pragma unroll
    for (int i = 0; i < 16; i++) {
        int off = i;                                       // {0..15}
        frag[i] = (__bf16)src[off];
    }
    (set ? Ai : Am)[(size_t)fragidx * 32 + lane] = frag;
}

// Pack normalized mesh into B tiles. Tile nt = 4KB contiguous:
//   [kb*1024 + lane*16)        : vals 0..7  (K = kb*32 + hb + 0..7)
//   [kb*1024 + 512 + lane*16)  : vals 8..15 (K = kb*32 + hb + 8..15), hb = (lane>=16)*16
// 16B per lane -> conflict-free ds_load_b128 in the GEMM.
__global__ __launch_bounds__(256) void k_packB(const float* __restrict__ nmm,
                                               const float* __restrict__ rnorm,
                                               v8bf* __restrict__ Bp) {
    int tid = blockIdx.x * blockDim.x + threadIdx.x;
    int fragidx = tid >> 5;                                // nt*4 + kb
    int lane    = tid & 31;
    int nt = fragidx >> 2, kb = fragidx & 3;
    int hb = (lane >> 4) * 16;
    int cj = nt * 16 + (lane & 15);
    float rn = rnorm[cj];
    const float* src = nmm + (size_t)cj * DD + kb * 32 + hb;
    v8bf lo, hh;
    #pragma unroll
    for (int i = 0; i < 8; i++) {
        lo[i] = (__bf16)(src[i] * rn);
        hh[i] = (__bf16)(src[8 + i] * rn);
    }
    size_t tb = (size_t)nt * 256 + kb * 64;                // tile = 256 x v8bf
    Bp[tb + lane]      = lo;
    Bp[tb + 32 + lane] = hh;
}

// TDM: DMA one 4KB B tile (1024 dwords, contiguous) from global into LDS.
__device__ __forceinline__ void tdm_load_b_tile(unsigned long long gaddr,
                                                unsigned int lds_off) {
    u32x4 g0;
    g0[0] = 1u;                                            // count=1, user desc
    g0[1] = lds_off;                                       // LDS byte address
    g0[2] = (unsigned int)(gaddr & 0xffffffffull);
    g0[3] = (unsigned int)((gaddr >> 32) & 0x01ffffffull) | (2u << 30); // type=2
    i32x8 g1;
    g1[0] = (int)(2u << 16);                               // data_size = 4B
    g1[1] = (int)(1024u << 16);                            // tensor_dim0 = 1024
    g1[2] = (int)(1u << 16);                               // tensor_dim1 = 1
    g1[3] = (int)(1024u << 16);                            // tile_dim0 = 1024 dw
    g1[4] = 0;                                             // tile_dim1/2 unused
    g1[5] = 1024;                                          // tensor_dim0_stride
    g1[6] = 0;
    g1[7] = 0;
    i32x4 z4 = {0, 0, 0, 0};
#if __has_include(<hip/amd_detail/amd_gfx1250_TDM.h>)
    i32x8 z8 = {0, 0, 0, 0, 0, 0, 0, 0};
    __builtin_amdgcn_tensor_load_to_lds(g0, g1, z4, z4, z8, 0);
#else
    __builtin_amdgcn_tensor_load_to_lds(g0, g1, z4, z4, 0);
#endif
}

// Fused GEMM + online logsumexp. Grid: 512 = (set:2, rowblock:64, chunk:4).
// Block: 8 waves; wave w owns rows of tile mt = rb*8+w for ALL 192 tiles of its
// N-chunk. B tiles staged to LDS by TDM (wave 0), double-buffered. Per-column
// pad bias (0 or -1e30*kappa) staged to LDS once so the hot loop has no global
// traffic at all.
__global__ __launch_bounds__(256) void k_gemm_lse(
        const v16bf* __restrict__ Am, const v16bf* __restrict__ Ai,
        const v8bf*  __restrict__ Bp,
        const unsigned char* __restrict__ pad,
        const int* __restrict__ label,
        float* __restrict__ Pm, float* __restrict__ Ps, float* __restrict__ Pt) {
    // [0,8192): two B-tile buffers (TDM targets LDS offsets 0/4096)
    // [8192, 8192+12288): per-column pad bias for this chunk (3072 floats)
    __shared__ char smem[8192 + TPC * 16 * 4];
    float* biasLds = (float*)(smem + 8192);

    int blk   = blockIdx.x;
    int set   = blk >> 8;
    int rem   = blk & 255;
    int rb    = rem >> 2;
    int chunk = rem & 3;
    int wave  = threadIdx.x >> 5;
    int lane  = threadIdx.x & 31;
    int laneN = lane & 15;
    int hi    = lane >> 4;
    int mt    = rb * 8 + wave;
    int nt0   = chunk * TPC;

    // Touch B buffers so the compiler sees them as written (TDM is invisible),
    // and stage the chunk's pad bias.
    #pragma unroll
    for (int k = threadIdx.x; k < 2048; k += 256) ((float*)smem)[k] = 0.0f;
    for (int k = threadIdx.x; k < TPC * 16; k += 256)
        biasLds[k] = pad[nt0 * 16 + k] ? PAD_BIAS : 0.0f;

    const v16bf* A = set ? Ai : Am;
    v16bf a0 = A[(size_t)(mt * 4 + 0) * 32 + lane];
    v16bf a1 = A[(size_t)(mt * 4 + 1) * 32 + lane];
    v16bf a2 = A[(size_t)(mt * 4 + 2) * 32 + lane];
    v16bf a3 = A[(size_t)(mt * 4 + 3) * 32 + lane];

    int rowbase = mt * 16 + hi * 8;
    int tgt[8];
    #pragma unroll
    for (int q = 0; q < 8; q++) {
        int row = rowbase + q;
        tgt[q] = label[row >> 10] * VV + (row & (VV - 1));
    }

    float m[8], s[8], t[8];
    #pragma unroll
    for (int q = 0; q < 8; q++) { m[q] = -3.0e38f; s[q] = 0.0f; t[q] = 0.0f; }

    unsigned long long bbase = (unsigned long long)(uintptr_t)Bp;

    __syncthreads();                                   // LDS init done
    if (wave == 0) {
        tdm_load_b_tile(bbase + (unsigned long long)nt0 * 4096ull, 0u);
        __builtin_amdgcn_s_wait_tensorcnt(0);
    }
    __syncthreads();                                   // tile 0 resident

    for (int i = 0; i < TPC; ++i) {
        int p = i & 1;
        bool more = (i + 1) < TPC;
        if (wave == 0 && more)                         // prefetch next tile
            tdm_load_b_tile(bbase + (unsigned long long)(nt0 + i + 1) * 4096ull,
                            (unsigned)((p ^ 1) * 4096));

        // Load all four B fragments into distinct registers so the 8
        // ds_load_b128 can clause and the WMMAs retire on partial dscnt waits.
        const v8bf* bl = (const v8bf*)(smem + p * 4096);
        v8bf l0 = bl[  0 + lane], h0 = bl[ 32 + lane];
        v8bf l1 = bl[ 64 + lane], h1 = bl[ 96 + lane];
        v8bf l2 = bl[128 + lane], h2 = bl[160 + lane];
        v8bf l3 = bl[192 + lane], h3 = bl[224 + lane];
        float bias = biasLds[i * 16 + laneN];

        v8f acc = {};
        acc = __builtin_amdgcn_wmma_f32_16x16x32_bf16(false, a0, false,
                SHUF16(l0, h0), (short)0, acc, false, false);
        acc = __builtin_amdgcn_wmma_f32_16x16x32_bf16(false, a1, false,
                SHUF16(l1, h1), (short)0, acc, false, false);
        acc = __builtin_amdgcn_wmma_f32_16x16x32_bf16(false, a2, false,
                SHUF16(l2, h2), (short)0, acc, false, false);
        acc = __builtin_amdgcn_wmma_f32_16x16x32_bf16(false, a3, false,
                SHUF16(l3, h3), (short)0, acc, false, false);

        int col = (nt0 + i) * 16 + laneN;
        #pragma unroll
        for (int q = 0; q < 8; ++q) {                  // branchless online LSE
            float x = fmaf(acc[q], KAPPA, bias);       // pad folded into bias
            t[q] += (tgt[q] == col) ? x : 0.0f;
            float nm = fmaxf(m[q], x);
            s[q] = s[q] * __expf(m[q] - nm) + __expf(x - nm);
            m[q] = nm;
        }

        if (wave == 0 && more) __builtin_amdgcn_s_wait_tensorcnt(0);
        __syncthreads();                               // next tile resident /
    }                                                  // old buffer reusable

    // Merge (m,s,t) across the 16 lanes sharing each row (halves disjoint).
    #pragma unroll
    for (int q = 0; q < 8; q++) {
        #pragma unroll
        for (int mask = 1; mask < 16; mask <<= 1) {
            float om = __shfl_xor(m[q], mask, 32);
            float os = __shfl_xor(s[q], mask, 32);
            float ot = __shfl_xor(t[q], mask, 32);
            float nm = fmaxf(m[q], om);
            s[q] = s[q] * __expf(m[q] - nm) + os * __expf(om - nm);
            m[q] = nm;
            t[q] += ot;
        }
    }
    if (laneN == 0) {
        #pragma unroll
        for (int q = 0; q < 8; q++) {
            int row = rowbase + q;
            int idx = (set * CHUNKS + chunk) * MROWS + row;
            Pm[idx] = m[q]; Ps[idx] = s[q]; Pt[idx] = t[q];
        }
    }
}

// Merge chunk partials -> CE -> masked sums. One thread per (set,row).
__global__ __launch_bounds__(256) void k_ce(const float* __restrict__ Pm,
                                            const float* __restrict__ Ps,
                                            const float* __restrict__ Pt,
                                            const unsigned char* __restrict__ vis_m,
                                            const unsigned char* __restrict__ vis_i,
                                            float* __restrict__ accum) {
    int tid = blockIdx.x * blockDim.x + threadIdx.x;   // 0..16383
    int set = tid >> 13;                               // uniform per block
    int row = tid & (MROWS - 1);
    float M = -3.0e38f, S = 0.0f, T = 0.0f;
    #pragma unroll
    for (int c = 0; c < CHUNKS; ++c) {
        int idx = (set * CHUNKS + c) * MROWS + row;
        float om = Pm[idx], os = Ps[idx];
        float nm = fmaxf(M, om);
        S = S * __expf(M - nm) + os * __expf(om - nm);
        M = nm;
        T += Pt[idx];
    }
    float vz = (float)((set ? vis_i : vis_m)[row]);
    float ce = (M + __logf(S) - T) * vz;
    #pragma unroll
    for (int mask = 1; mask < 32; mask <<= 1) {
        ce += __shfl_xor(ce, mask, 32);
        vz += __shfl_xor(vz, mask, 32);
    }
    __shared__ float Rc[8], Rv[8];
    int wave = threadIdx.x >> 5, lane = threadIdx.x & 31;
    if (lane == 0) { Rc[wave] = ce; Rv[wave] = vz; }
    __syncthreads();
    if (threadIdx.x == 0) {
        float tc = 0.0f, tv = 0.0f;
        #pragma unroll
        for (int w = 0; w < 8; w++) { tc += Rc[w]; tv += Rv[w]; }
        atomicAdd(&accum[set * 2 + 0], tc);
        atomicAdd(&accum[set * 2 + 1], tv);
    }
}

__global__ __launch_bounds__(256) void k_bg(const float* __restrict__ bg_m,
                                            const float* __restrict__ bg_i,
                                            const float* __restrict__ gt_m,
                                            const float* __restrict__ gt_i,
                                            float* __restrict__ accum, int n) {
    float sm = 0.0f, si = 0.0f;
    for (int idx = blockIdx.x * blockDim.x + threadIdx.x; idx < n;
         idx += gridDim.x * blockDim.x) {
        float dm = bg_m[idx] - gt_m[idx]; sm += dm * dm;
        float di = bg_i[idx] - gt_i[idx]; si += di * di;
    }
    #pragma unroll
    for (int mask = 1; mask < 32; mask <<= 1) {
        sm += __shfl_xor(sm, mask, 32);
        si += __shfl_xor(si, mask, 32);
    }
    __shared__ float Wm[8], Wi[8];
    int wave = threadIdx.x >> 5, lane = threadIdx.x & 31;
    if (lane == 0) { Wm[wave] = sm; Wi[wave] = si; }
    __syncthreads();
    if (threadIdx.x == 0) {
        float tm = 0.0f, ti = 0.0f;
        #pragma unroll
        for (int w = 0; w < 8; w++) { tm += Wm[w]; ti += Wi[w]; }
        atomicAdd(&accum[4], tm);
        atomicAdd(&accum[5], ti);
    }
}

__global__ void k_final(const float* __restrict__ accum, float* __restrict__ out) {
    if (threadIdx.x == 0 && blockIdx.x == 0) {
        float loss_m = accum[0] / accum[1];
        float loss_i = accum[2] / accum[3];
        out[0] = 0.5f * (loss_m + loss_i);
        out[1] = 0.5f * (accum[4] + accum[5]) * (1.0f / (float)BG_N);
    }
}

// ---------------------------------------------------------------------------
extern "C" void kernel_launch(void* const* d_in, const int* in_sizes, int n_in,
                              void* d_out, int out_size, void* d_ws, size_t ws_size,
                              hipStream_t stream) {
    (void)in_sizes; (void)n_in; (void)out_size; (void)ws_size;
    const float*         kp_m  = (const float*)d_in[0];
    const float*         kp_i  = (const float*)d_in[1];
    const int*           label = (const int*)d_in[2];
    const unsigned char* vis_m = (const unsigned char*)d_in[3];
    const unsigned char* vis_i = (const unsigned char*)d_in[4];
    const float*         nmm   = (const float*)d_in[5];
    const unsigned char* pad   = (const unsigned char*)d_in[6];
    const float*         bg_m  = (const float*)d_in[7];
    const float*         bg_i  = (const float*)d_in[8];
    const float*         gt_m  = (const float*)d_in[9];
    const float*         gt_i  = (const float*)d_in[10];
    float* out = (float*)d_out;

    char*  ws    = (char*)d_ws;
    float* accum = (float*)ws;
    float* rnorm = accum + 16;
    float* Pm = (float*)(ws + (64 << 10));
    float* Ps = (float*)(ws + (64 << 10) + (256 << 10));
    float* Pt = (float*)(ws + (64 << 10) + (512 << 10));
    v16bf* Am = (v16bf*)(ws + (1 << 20));
    v16bf* Ai = (v16bf*)(ws + (3 << 20));
    v8bf*  Bp = (v8bf*) (ws + (5 << 20));

    k_init_accum<<<1, 32, 0, stream>>>(accum);
    k_norm <<<NCOLS * 32 / 256, 256, 0, stream>>>(nmm, rnorm);
    k_packA<<<2 * MT * KB * 32 / 256, 256, 0, stream>>>(kp_m, kp_i, Am, Ai);
    k_packB<<<NT * KB * 32 / 256, 256, 0, stream>>>(nmm, rnorm, Bp);
    k_gemm_lse<<<2 * 64 * CHUNKS, 256, 0, stream>>>(Am, Ai, Bp, pad, label,
                                                    Pm, Ps, Pt);
    k_ce   <<<2 * MROWS / 256, 256, 0, stream>>>(Pm, Ps, Pt, vis_m, vis_i, accum);
    k_bg   <<<512, 256, 0, stream>>>(bg_m, bg_i, gt_m, gt_i, accum, BG_N);
    k_final<<<1, 1, 0, stream>>>(accum, out);
}
